// SRFTSketchOp_53163105190180
// MI455X (gfx1250) — compile-verified, matrix operand order checked by reading, not run on hardware
//
#include <hip/hip_runtime.h>
#include <hip/hip_bf16.h>

// SRFT sketch as a fused GEMM:
//   out[d,m] = (1/N) * sum_k A[d,k] * (D[k]*M[k,m])
//   A[d,k]   = cos(pi*(2*P[d]+1)*k / (2N)),  A[d,0] folded to 0.5
// A generated in registers (exact integer phase mod 4N = 2^15, v_cos_f32,
// strength-reduced phase recurrence). B staged through double-buffered LDS in
// f16. All 8 B fragments preloaded into distinct registers -> one s_wait_dscnt
// per K-step, WMMAs issue back-to-back while next tile's global loads fly.
// Core compute: V_WMMA_F32_16X16X32_F16, 8 WMMAs amortize each A fragment.

typedef __attribute__((ext_vector_type(16))) _Float16 v16h;
typedef __attribute__((ext_vector_type(8)))  float    v8f;

#define N_DIM   8192
#define D_SK    1024
#define MCOLS   4096
#define BK      32       // K tile (one WMMA depth)
#define BM      128      // d-rows per block (8 waves x 16-row strips)
#define BN      128      // m-cols per block (each wave covers all 128)
#define NFRAG   8        // 16x16 output fragments per wave
#define LDS_KS  (BK + 8) // padded k-stride in halves: 40 -> 80B lane stride, conflict-free

typedef union { v16h v; unsigned u[8]; } bfrag_t;

__global__ __launch_bounds__(256)
void srft_sketch_wmma(const float* __restrict__ Mg,
                      const float* __restrict__ Dg,
                      const int*   __restrict__ Pg,
                      float*       __restrict__ out)
{
    // Double-buffered B tile, transposed: Bs[buf][n][k], f16, padded
    __shared__ _Float16 Bs[2][BN][LDS_KS];   // 2*128*40*2 = 20 KB

    const int tid     = threadIdx.x;
    const int lane    = tid & 31;
    const int wave    = tid >> 5;         // 0..7 -> 16-row strip
    const int half    = lane >> 4;        // K-half selector (ISA 16-bit layout)
    const int lr      = lane & 15;        // row (A) / col (B,C) within fragment

    const int row0     = blockIdx.y * BM + wave * 16;  // output d-rows of this wave
    const int colBlock = blockIdx.x * BN;              // block m-col base

    // Per-lane row phase constant: c = 2*P[row]+1 (both lane halves share rows)
    const unsigned cmul = 2u * (unsigned)Pg[row0 + lr] + 1u;

    // Loop-invariant phase bases: cbase[h] = cmul * koff(h), koff = (h&7)+8*half+16*(h>>3)
    unsigned cbase[16];
    #pragma unroll
    for (int h = 0; h < 16; ++h) {
        const unsigned koff = (unsigned)((h & 7) + 8 * half + 16 * (h >> 3));
        cbase[h] = cmul * koff;
    }
    const unsigned cstep = cmul * (unsigned)BK;  // phase increment per K-step
    unsigned phase = 0;                          // cmul * k0 (mod 2^32)

    v8f acc[NFRAG];
    #pragma unroll
    for (int j = 0; j < NFRAG; ++j) acc[j] = (v8f){};

    // Staging map: 256 threads load a 32(k) x 128(n) f32 tile of M, coalesced.
    // 8 threads per k-row, each loads 4 x float4 (16 floats).
    const int srow = tid >> 3;            // k within tile: 0..31
    const int sgrp = (tid & 7) * 16;      // n offset: 0..112

    const int nsteps = N_DIM / BK;        // 256

    // ---- prologue: stage tile 0 into buffer 0 ----
    {
        const float dsign = Dg[srow];
        const float* mrow = Mg + (size_t)srow * MCOLS + colBlock + sgrp;
        #pragma unroll
        for (int q = 0; q < 4; ++q) {
            const float4 v = *(const float4*)(mrow + 4 * q);
            const int nb = sgrp + 4 * q;
            Bs[0][nb + 0][srow] = (_Float16)(v.x * dsign);
            Bs[0][nb + 1][srow] = (_Float16)(v.y * dsign);
            Bs[0][nb + 2][srow] = (_Float16)(v.z * dsign);
            Bs[0][nb + 3][srow] = (_Float16)(v.w * dsign);
        }
    }
    __syncthreads();

    for (int s = 0; s < nsteps; ++s) {
        const int k0  = s * BK;
        const int cur = s & 1;
        const int nxt = cur ^ 1;
        const bool havenext = (s + 1 < nsteps);

        // ---- issue next tile's global loads first (in flight during WMMAs) ----
        float4 sv[4];
        float  dsign = 0.0f;
        if (havenext) {
            const int kn = k0 + BK;
            dsign = Dg[kn + srow];
            const float* mrow = Mg + (size_t)(kn + srow) * MCOLS + colBlock + sgrp;
            if (s + 2 < nsteps) {
                __builtin_prefetch(mrow + (size_t)BK * MCOLS, 0, 1); // global_prefetch_b8
            }
            #pragma unroll
            for (int q = 0; q < 4; ++q) sv[q] = *(const float4*)(mrow + 4 * q);
        }

        // ---- preload ALL 8 B fragments into distinct registers ----
        const _Float16* bb = &Bs[cur][0][0];
        const int kkbase = 8 * half;                   // K pair base for this lane half
        bfrag_t bfr[NFRAG];
        #pragma unroll
        for (int j = 0; j < NFRAG; ++j) {
            const _Float16* bp = bb + (size_t)(j * 16 + lr) * LDS_KS + kkbase;
            #pragma unroll
            for (int q = 0; q < 8; ++q) {
                const int kk = 2 * (q & 3) + 16 * (q >> 2);
                bfr[j].u[q] = *(const unsigned*)(bp + kk);
            }
        }

        // ---- generate A fragment in registers (16x32 f16, ISA layout) ----
        v16h afrag;
        #pragma unroll
        for (int h = 0; h < 16; ++h) {
            const unsigned t = (phase + cbase[h]) & 32767u;       // exact phase mod 4N
            float val = __builtin_amdgcn_cosf((float)t * (1.0f / 32768.0f));
            if (h == 0) {
                // k==0 carries the DCT-III half weight; only h==0/half==0 can hit it
                if ((k0 == 0) && (half == 0)) val = 0.5f;
            }
            afrag[h] = (_Float16)val;
        }
        phase += cstep;

        // ---- 8 back-to-back WMMAs (single dscnt wait covers all loads) ----
        #pragma unroll
        for (int j = 0; j < NFRAG; ++j) {
            acc[j] = __builtin_amdgcn_wmma_f32_16x16x32_f16(
                /*neg_a=*/false, afrag, /*neg_b=*/false, bfr[j].v,
                /*c_mod=*/(short)0, acc[j], /*reuse_a=*/false, /*reuse_b=*/false);
        }

        // ---- write staged registers into the other LDS buffer ----
        if (havenext) {
            #pragma unroll
            for (int q = 0; q < 4; ++q) {
                const int nb = sgrp + 4 * q;
                Bs[nxt][nb + 0][srow] = (_Float16)(sv[q].x * dsign);
                Bs[nxt][nb + 1][srow] = (_Float16)(sv[q].y * dsign);
                Bs[nxt][nb + 2][srow] = (_Float16)(sv[q].z * dsign);
                Bs[nxt][nb + 3][srow] = (_Float16)(sv[q].w * dsign);
            }
        }
        __syncthreads();
    }

    // ---- store: f32 C/D layout (VGPR q -> rows q and q+8; lane%16 -> col) ----
    const float scale = 1.0f / (float)N_DIM;
    #pragma unroll
    for (int j = 0; j < NFRAG; ++j) {
        const int col = colBlock + j * 16 + lr;
        #pragma unroll
        for (int q = 0; q < 8; ++q) {
            const int r = row0 + q + 8 * half;
            out[(size_t)r * MCOLS + col] = acc[j][q] * scale;
        }
    }
}

extern "C" void kernel_launch(void* const* d_in, const int* in_sizes, int n_in,
                              void* d_out, int out_size, void* d_ws, size_t ws_size,
                              hipStream_t stream) {
    (void)in_sizes; (void)n_in; (void)d_ws; (void)ws_size; (void)out_size;
    const float* Mg = (const float*)d_in[0];   // [8192, 4096] f32
    const float* Dg = (const float*)d_in[1];   // [8192] f32 (Rademacher)
    const int*   Pg = (const int*)d_in[2];     // [1024] int32 row picks
    float*       og = (float*)d_out;           // [1024, 4096] f32

    dim3 grid(MCOLS / BN, D_SK / BM);          // (32, 8) = 256 blocks
    dim3 block(256);                           // 8 wave32 waves
    srft_sketch_wmma<<<grid, block, 0, stream>>>(Mg, Dg, Pg, og);
}
